// TFLayoutLMv3SelfAttention_41712722378940
// MI455X (gfx1250) — matrix operand order repaired and checked by reading, not compiled, not run on hardware
//
#include <hip/hip_runtime.h>

typedef __attribute__((ext_vector_type(16))) _Float16 v16h;
typedef __attribute__((ext_vector_type(8)))  _Float16 v8h;
typedef __attribute__((ext_vector_type(8)))  float    v8f;

#define WMMA(a, b, c) __builtin_amdgcn_wmma_f32_16x16x32_f16(false, (a), false, (b), (short)0, (c), false, false)

constexpr int B_  = 4;
constexpr int S_  = 1024;
constexpr int H_  = 768;
constexpr int NH_ = 12;
constexpr int HD_ = 64;
constexpr float INV_NORM = 0.125f;   // 1/sqrt(64)

union V16 { v16h v; v8h h2[2]; };

// ---------------------------------------------------------------------------
// Kernel 1: hidden_states f32 -> f16
// ---------------------------------------------------------------------------
__global__ void k_cvt_x(const float* __restrict__ x, _Float16* __restrict__ y, int n) {
  int i = blockIdx.x * blockDim.x + threadIdx.x;
  if (i < n) y[i] = (_Float16)x[i];
}

// ---------------------------------------------------------------------------
// Kernel 2: W[k][n] f32 -> Wt[n][k] f16 (LDS tiled transpose), z = proj
// ---------------------------------------------------------------------------
__global__ void k_cvt_wt(const float* __restrict__ Wq, const float* __restrict__ Wk,
                         const float* __restrict__ Wv,
                         _Float16* __restrict__ Wtq, _Float16* __restrict__ Wtk,
                         _Float16* __restrict__ Wtv) {
  __shared__ float tile[32][33];
  const float* W  = (blockIdx.z == 0) ? Wq  : (blockIdx.z == 1) ? Wk  : Wv;
  _Float16*   Wt  = (blockIdx.z == 0) ? Wtq : (blockIdx.z == 1) ? Wtk : Wtv;
  const int tx = threadIdx.x, ty = threadIdx.y;       // 32 x 8
  const int x0 = blockIdx.x * 32, y0 = blockIdx.y * 32;
  #pragma unroll
  for (int j = 0; j < 32; j += 8)
    tile[ty + j][tx] = W[(size_t)(y0 + ty + j) * H_ + x0 + tx];
  __syncthreads();
  #pragma unroll
  for (int j = 0; j < 32; j += 8)
    Wt[(size_t)(x0 + ty + j) * H_ + y0 + tx] = (_Float16)tile[tx][ty + j];
}

// ---------------------------------------------------------------------------
// Kernel 3: QKV projections. One wave computes a 16x64 output tile.
//   proj 0/1 (Q,K): C = X (16x768) * W (768x64), stored [BH, S, HD] f16
//                   (Q pre-scaled by 1/sqrt(HD))
//   proj 2   (V)  : computed transposed, C = Wt (64x768) * X^T (768x16),
//                   stored [BH, HD, S] f16 -> contiguous B-frags in attention
// ---------------------------------------------------------------------------
__global__ void __launch_bounds__(32) k_qkv(
    const _Float16* __restrict__ Xh,
    const _Float16* __restrict__ Wtq, const _Float16* __restrict__ Wtk,
    const _Float16* __restrict__ Wtv,
    const float* __restrict__ bq, const float* __restrict__ bk,
    const float* __restrict__ bv,
    _Float16* __restrict__ qh, _Float16* __restrict__ kh,
    _Float16* __restrict__ vt) {
  const int lane   = threadIdx.x;
  const int col    = lane & 15;
  const int hi     = lane >> 4;
  const int base8  = hi * 8;
  const int base16 = hi * 16;
  const int mt     = blockIdx.x;   // 0..255  (16-row block of the 4096 rows)
  const int h      = blockIdx.y;   // 0..11   (head == 64-col block)
  const int proj   = blockIdx.z;   // 0=Q 1=K 2=V

  const _Float16* Wt = (proj == 0) ? Wtq : (proj == 1) ? Wtk : Wtv;
  v8f acc[4] = {};

  if (proj < 2) {
    for (int c = 0; c < H_; c += 32) {
      V16 a;
      const _Float16* xrow = Xh + (size_t)(mt * 16 + col) * H_ + c;
      a.h2[0] = *(const v8h*)(xrow + base16);
      a.h2[1] = *(const v8h*)(xrow + base16 + 8);
      #pragma unroll
      for (int nt = 0; nt < 4; ++nt) {
        V16 bf;
        const _Float16* wrow = Wt + (size_t)(h * 64 + nt * 16 + col) * H_ + c + base16;
        bf.h2[0] = *(const v8h*)(wrow);
        bf.h2[1] = *(const v8h*)(wrow + 8);
        acc[nt] = WMMA(a.v, bf.v, acc[nt]);
      }
    }
    const float* bias  = (proj == 0) ? bq : bk;
    _Float16*    outp  = (proj == 0) ? qh : kh;
    const float  scale = (proj == 0) ? INV_NORM : 1.0f;
    #pragma unroll
    for (int nt = 0; nt < 4; ++nt) {
      const float bn = bias[h * 64 + nt * 16 + col];
      #pragma unroll
      for (int r = 0; r < 8; ++r) {
        const int Mg = mt * 16 + base8 + r;
        const int b  = Mg >> 10, s = Mg & 1023;
        outp[((size_t)(b * NH_ + h) * S_ + s) * HD_ + nt * 16 + col] =
            (_Float16)((acc[nt][r] + bn) * scale);
      }
    }
  } else {
    for (int c = 0; c < H_; c += 32) {
      V16 bf;   // B = X^T chunk: contraction along k, column = global row m
      const _Float16* xrow = Xh + (size_t)(mt * 16 + col) * H_ + c + base16;
      bf.h2[0] = *(const v8h*)(xrow);
      bf.h2[1] = *(const v8h*)(xrow + 8);
      #pragma unroll
      for (int nt = 0; nt < 4; ++nt) {
        V16 a;  // A = Wt rows (output channels)
        const _Float16* wrow = Wt + (size_t)(h * 64 + nt * 16 + col) * H_ + c;
        a.h2[0] = *(const v8h*)(wrow + base16);
        a.h2[1] = *(const v8h*)(wrow + base16 + 8);
        acc[nt] = WMMA(a.v, bf.v, acc[nt]);
      }
    }
    const int b  = mt >> 6;
    const int s0 = (mt & 63) * 16;
    #pragma unroll
    for (int nt = 0; nt < 4; ++nt) {
      #pragma unroll
      for (int r = 0; r < 8; ++r) {
        const int   d  = nt * 16 + base8 + r;
        const float bn = bv[h * 64 + d];
        vt[((size_t)(b * NH_ + h) * HD_ + d) * S_ + s0 + col] =
            (_Float16)(acc[nt][r] + bn);
      }
    }
  }
}

// ---------------------------------------------------------------------------
// Kernel 4: fused attention (flash-style online softmax).
// One wave handles 16 query rows of one (b,h); streams 32-key blocks.
// PB-Relax softmax == standard max-subtracted softmax, computed exactly.
// ---------------------------------------------------------------------------
__global__ void __launch_bounds__(32) k_attn(
    const _Float16* __restrict__ qh, const _Float16* __restrict__ kh,
    const _Float16* __restrict__ vt,
    const float* __restrict__ rel1, const float* __restrict__ rel2,
    const float* __restrict__ mask, float* __restrict__ out) {
  __shared__ _Float16 pl[16 * 32];     // P tile staging (C-layout -> A-layout)
  const int lane   = threadIdx.x;
  const int col    = lane & 15;
  const int hi     = lane >> 4;
  const int base8  = hi * 8;
  const int base16 = hi * 16;
  const int qt     = blockIdx.x;       // 0..63
  const int bh     = blockIdx.y;       // 0..47
  const int b      = bh / NH_;
  const int h      = bh - b * NH_;

  // Q A-fragments (d = 0..31 and 32..63); Q already scaled by 1/sqrt(HD)
  V16 qa0, qa1;
  {
    const _Float16* qrow = qh + ((size_t)bh * S_ + qt * 16 + col) * HD_;
    qa0.h2[0] = *(const v8h*)(qrow + base16);
    qa0.h2[1] = *(const v8h*)(qrow + base16 + 8);
    qa1.h2[0] = *(const v8h*)(qrow + 32 + base16);
    qa1.h2[1] = *(const v8h*)(qrow + 40 + base16);
  }

  v8f o[4] = {};
  float m8[8], l8[8];
  #pragma unroll
  for (int r = 0; r < 8; ++r) { m8[r] = -1e30f; l8[r] = 0.f; }

  const size_t relbase = ((size_t)bh * S_ + qt * 16) * S_;
  const float* maskb   = mask + (size_t)b * S_;

  for (int kt = 0; kt < 32; ++kt) {
    const int key0 = kt * 32;

    // ---- scores: two 16x16 tiles (keys key0..+15 and +16..+31) ----
    v8f s0 = {}, s1 = {};
    {
      V16 kb;
      const _Float16* krow = kh + ((size_t)bh * S_ + key0 + col) * HD_ + base16;
      kb.h2[0] = *(const v8h*)(krow);      kb.h2[1] = *(const v8h*)(krow + 8);
      s0 = WMMA(qa0.v, kb.v, s0);
      kb.h2[0] = *(const v8h*)(krow + 32); kb.h2[1] = *(const v8h*)(krow + 40);
      s0 = WMMA(qa1.v, kb.v, s0);
      const _Float16* krow1 = krow + 16 * HD_;
      kb.h2[0] = *(const v8h*)(krow1);      kb.h2[1] = *(const v8h*)(krow1 + 8);
      s1 = WMMA(qa0.v, kb.v, s1);
      kb.h2[0] = *(const v8h*)(krow1 + 32); kb.h2[1] = *(const v8h*)(krow1 + 40);
      s1 = WMMA(qa1.v, kb.v, s1);
    }

    // prefetch next iteration's rel rows (global_prefetch_b8)
    if (kt + 1 < 32) {
      const float* pf = ((hi == 0) ? rel1 : rel2) + relbase + (size_t)col * S_ + key0 + 32;
      __builtin_prefetch(pf, 0, 0);
    }

    // ---- bias terms: (rel_pos + rel_2d_pos)/norm + mask ----
    const float mask0 = maskb[key0 + col];
    const float mask1 = maskb[key0 + 16 + col];
    #pragma unroll
    for (int r = 0; r < 8; ++r) {
      const size_t idx = relbase + (size_t)(base8 + r) * S_ + key0 + col;
      s0[r] += (rel1[idx]      + rel2[idx])      * INV_NORM + mask0;
      s1[r] += (rel1[idx + 16] + rel2[idx + 16]) * INV_NORM + mask1;
    }

    // ---- online softmax update (row stats replicated in 16-lane groups) ----
    float alpha[8], p0[8], p1[8];
    #pragma unroll
    for (int r = 0; r < 8; ++r) {
      float t = fmaxf(s0[r], s1[r]);
      t = fmaxf(t, __shfl_xor(t, 1));
      t = fmaxf(t, __shfl_xor(t, 2));
      t = fmaxf(t, __shfl_xor(t, 4));
      t = fmaxf(t, __shfl_xor(t, 8));
      const float mn = fmaxf(m8[r], t);
      alpha[r] = __expf(m8[r] - mn);
      m8[r]    = mn;
      p0[r]    = __expf(s0[r] - mn);
      p1[r]    = __expf(s1[r] - mn);
      float rs = p0[r] + p1[r];
      rs += __shfl_xor(rs, 1);
      rs += __shfl_xor(rs, 2);
      rs += __shfl_xor(rs, 4);
      rs += __shfl_xor(rs, 8);
      l8[r] = l8[r] * alpha[r] + rs;
      o[0][r] *= alpha[r];
      o[1][r] *= alpha[r];
      o[2][r] *= alpha[r];
      o[3][r] *= alpha[r];
    }

    // ---- stage P (f16) through LDS: C-layout -> A-layout (in-order per wave)
    #pragma unroll
    for (int r = 0; r < 8; ++r) {
      pl[(base8 + r) * 32 + col]      = (_Float16)p0[r];
      pl[(base8 + r) * 32 + 16 + col] = (_Float16)p1[r];
    }
    V16 pa;
    pa.h2[0] = *(const v8h*)&pl[col * 32 + base16];
    pa.h2[1] = *(const v8h*)&pl[col * 32 + base16 + 8];

    // ---- PV: O += P(16x32) * V(32x64), V stored transposed [HD, S] ----
    #pragma unroll
    for (int nt = 0; nt < 4; ++nt) {
      V16 vb;
      const _Float16* vrow =
          vt + ((size_t)bh * HD_ + nt * 16 + col) * S_ + key0 + base16;
      vb.h2[0] = *(const v8h*)(vrow);
      vb.h2[1] = *(const v8h*)(vrow + 8);
      o[nt] = WMMA(pa.v, vb.v, o[nt]);
    }
  }

  // ---- normalize and write ctx: out[b, s, h*64 + d] (fp32) ----
  #pragma unroll
  for (int r = 0; r < 8; ++r) {
    const float inv  = 1.0f / l8[r];
    const int   srow = qt * 16 + base8 + r;
    float* orow = out + ((size_t)b * S_ + srow) * H_ + h * HD_;
    orow[ 0 + col] = o[0][r] * inv;
    orow[16 + col] = o[1][r] * inv;
    orow[32 + col] = o[2][r] * inv;
    orow[48 + col] = o[3][r] * inv;
  }
}

// ---------------------------------------------------------------------------
extern "C" void kernel_launch(void* const* d_in, const int* in_sizes, int n_in,
                              void* d_out, int out_size, void* d_ws, size_t ws_size,
                              hipStream_t stream) {
  (void)in_sizes; (void)n_in; (void)out_size; (void)ws_size;
  const float* hs   = (const float*)d_in[0];
  const float* mask = (const float*)d_in[1];
  const float* rel1 = (const float*)d_in[2];
  const float* rel2 = (const float*)d_in[3];
  const float* Wq   = (const float*)d_in[4];
  const float* bq   = (const float*)d_in[5];
  const float* Wk   = (const float*)d_in[6];
  const float* bk   = (const float*)d_in[7];
  const float* Wv   = (const float*)d_in[8];
  const float* bv   = (const float*)d_in[9];
  float* out = (float*)d_out;

  // workspace layout (f16 scratch, ~27.4 MB total)
  char* ws = (char*)d_ws;
  const size_t nX  = (size_t)B_ * S_ * H_;      // 3,145,728
  const size_t nW  = (size_t)H_ * H_;           //   589,824
  const size_t nQ  = (size_t)B_ * NH_ * S_ * HD_;
  _Float16* Xh  = (_Float16*)ws;
  _Float16* Wtq = Xh + nX;
  _Float16* Wtk = Wtq + nW;
  _Float16* Wtv = Wtk + nW;
  _Float16* qhp = Wtv + nW;
  _Float16* khp = qhp + nQ;
  _Float16* vtp = khp + nQ;

  k_cvt_x<<<(int)(nX / 256), 256, 0, stream>>>(hs, Xh, (int)nX);

  dim3 gw(H_ / 32, H_ / 32, 3), bw(32, 8);
  k_cvt_wt<<<gw, bw, 0, stream>>>(Wq, Wk, Wv, Wtq, Wtk, Wtv);

  dim3 gg(256, NH_, 3);
  k_qkv<<<gg, 32, 0, stream>>>(Xh, Wtq, Wtk, Wtv, bq, bk, bv, qhp, khp, vtp);

  dim3 ga(S_ / 16, B_ * NH_);
  k_attn<<<ga, 32, 0, stream>>>(qhp, khp, vtp, rel1, rel2, mask, out);
}
